// BiLSTM_CRF_72645076844862
// MI455X (gfx1250) — matrix operand order repaired and checked by reading, not compiled, not run on hardware
//
#include <hip/hip_runtime.h>

// ---------------------------------------------------------------------------
// BiLSTM-CRF NLL on gfx1250 (MI455X).
//   V=50000 E=128 HID=256 (H=128/dir) C=9 B=256 L=512
// Phases:
//   1) weight fp32->bf16 conversion (W_ih_f/b, W_hh_f/b)
//   2) embedding gather -> time-major bf16 (L,B,E)
//   3) input-gate GEMM per direction: (131072,128)x(128,512)+b, WMMA bf16;
//      output stored in *fragment order* ([tile][lane][8] bf16) so phase 4
//      reloads accumulators with one b128 per lane.
//   4) persistent-workgroup LSTM recurrence per direction (grid=2), WMMA,
//      h + W_hh in LDS, c in VGPRs, next-step gate prefetch, hs -> fp32
//   5) emissions (L,B,9) fp32 (VALU, tiny)
//   6) CRF numerator + forward algorithm + reduction -> scalar -sum(llh)
// ---------------------------------------------------------------------------

typedef __attribute__((ext_vector_type(16))) __bf16 v16bf;
typedef __attribute__((ext_vector_type(8)))  float  v8f;
typedef unsigned short u16;
typedef unsigned int   u32;

#define LEN   512
#define BATCH 256
#define EDIM  128
#define HDIM  128
#define GDIM  512   // 4*H
#define NTAG  9

union FragBF { uint4 q[2]; v16bf v; };

__device__ __forceinline__ u16 f2bf(float f) {
  u32 u = __float_as_uint(f);
  u += 0x7FFFu + ((u >> 16) & 1u);          // round-to-nearest-even
  return (u16)(u >> 16);
}
__device__ __forceinline__ float bf2f(u16 s) {
  return __uint_as_float(((u32)s) << 16);
}
__device__ __forceinline__ u32 pack2(float lo, float hi) {
  return (u32)f2bf(lo) | ((u32)f2bf(hi) << 16);
}
// Load a 16-element bf16 fragment as two 16B chunks (must be 16B aligned).
__device__ __forceinline__ v16bf load_frag(const u16* p0, const u16* p1) {
  FragBF f;
  f.q[0] = *(const uint4*)p0;
  f.q[1] = *(const uint4*)p1;
  return f.v;
}
__device__ __forceinline__ float sigm(float x) { return 1.0f / (1.0f + expf(-x)); }

// ---------------------------------------------------------------------------
// Phase 1: fp32 -> bf16 conversion
// ---------------------------------------------------------------------------
__global__ void cvt_bf16_kernel(const float* __restrict__ src, u16* __restrict__ dst, int n) {
  int i = blockIdx.x * blockDim.x + threadIdx.x;
  if (i < n) dst[i] = f2bf(src[i]);
}

// ---------------------------------------------------------------------------
// Phase 2: embedding gather -> time-major (L, B, E) bf16
// ---------------------------------------------------------------------------
__global__ void embed_kernel(const int* __restrict__ x, const float* __restrict__ emb,
                             u16* __restrict__ out) {
  int idx = blockIdx.x * blockDim.x + threadIdx.x;   // over L*B*E = 16777216
  int l   = idx >> 15;             // / (B*E)
  int rem = idx & 32767;
  int b   = rem >> 7;
  int k   = rem & 127;
  int id  = x[b * LEN + l];
  out[idx] = f2bf(emb[(size_t)id * EDIM + k]);
}

// ---------------------------------------------------------------------------
// Phase 3: input-projection GEMM: bias[n] + sum_k X[m][k]*W[n][k].
// One 16x16 tile per wave, K=128 = 4 x wmma_f32_16x16x32_bf16.
// Output in fragment order: Gout[((mt*32+ntg)*32 + lane)*8 + r] (bf16),
// i.e. one uint4 (8 bf16) per lane per tile -> single b128 store.
// ---------------------------------------------------------------------------
__global__ __launch_bounds__(256)
void gemm_gates_kernel(const u16* __restrict__ Amat, const u16* __restrict__ Wbf,
                       const float* __restrict__ bias, u16* __restrict__ Gout) {
  int wid  = blockIdx.x * 8 + (threadIdx.x >> 5);   // 262144 wave-tiles
  int lane = threadIdx.x & 31;
  int mt   = wid >> 5;                              // 0..8191  (M-tiles over L*B)
  int ntg  = wid & 31;                              // 0..31    (gate-col tiles)
  int lm   = lane & 15;
  int hi   = lane >> 4;

  const u16* arow = Amat + (size_t)(mt * 16 + lm) * EDIM;
  const u16* brow = Wbf  + (size_t)(ntg * 16 + lm) * EDIM;
  float bv = bias[ntg * 16 + lm];
  v8f acc;
#pragma unroll
  for (int r = 0; r < 8; ++r) acc[r] = bv;

#pragma unroll
  for (int ks = 0; ks < 4; ++ks) {
    int ao = ks * 32 + (hi << 3);                   // A: lanes 0-15 K{0..7,16..23}; 16-31 K{8..15,24..31}
    v16bf a = load_frag(arow + ao, arow + ao + 16);
    int bo = ks * 32 + (hi << 3);                   // B: lanes 0-15 K{0..7,16..23}; 16-31 K{8..15,24..31}
    v16bf b = load_frag(brow + bo, brow + bo + 16);
    acc = __builtin_amdgcn_wmma_f32_16x16x32_bf16(false, a, false, b, (short)0, acc, false, false);
  }

  uint4 pk;
  pk.x = pack2(acc[0], acc[1]);
  pk.y = pack2(acc[2], acc[3]);
  pk.z = pack2(acc[4], acc[5]);
  pk.w = pack2(acc[6], acc[7]);
  *(uint4*)(Gout + (((size_t)mt * 32 + ntg) * 32 + lane) * 8) = pk;
}

// ---------------------------------------------------------------------------
// Phase 4: LSTM recurrence. grid=2 (blockIdx.x = direction), 1024 threads.
// LDS: W_hh bf16 (128KB) + h bf16 (64KB). c state in VGPRs.
// Wave w: M-tile mt=w>>1, 4 h-tiles at N-tiles (w&1)*4 .. +3; each h-tile
// uses gate tiles i/f/g/o -> nonlinearity entirely in registers.
// Gate accumulators init from fragment-ordered gates: one b128 per lane.
// Next step's fragments prefetched (global_prefetch_b8) each iteration.
// ---------------------------------------------------------------------------
__global__ __launch_bounds__(1024)
void lstm_rec_kernel(const u16* __restrict__ whh_f, const u16* __restrict__ whh_b,
                     const u16* __restrict__ gates_f, const u16* __restrict__ gates_b,
                     float* __restrict__ hs_f, float* __restrict__ hs_b) {
  extern __shared__ u16 smem[];
  u16* wlds = smem;                 // 512*128 bf16
  u16* hlds = smem + GDIM * HDIM;   // 256*128 bf16

  const int dir = blockIdx.x;
  const u16* whh   = dir ? whh_b   : whh_f;
  const u16* gates = dir ? gates_b : gates_f;
  float*     hs    = dir ? hs_b    : hs_f;

  int tid = threadIdx.x;
  { // stage W_hh into LDS, zero h
    const u32* wsrc = (const u32*)whh;
    u32* wdst = (u32*)wlds;
    for (int i = tid; i < GDIM * HDIM / 2; i += 1024) wdst[i] = wsrc[i];
    u32* hdst = (u32*)hlds;
    for (int i = tid; i < BATCH * HDIM / 2; i += 1024) hdst[i] = 0u;
  }
  __syncthreads();

  int wid = tid >> 5, lane = tid & 31, lm = lane & 15, hi = lane >> 4;
  int mt  = wid >> 1;
  int ntb = (wid & 1) * 4;

  v8f cs[4];
#pragma unroll
  for (int j = 0; j < 4; ++j)
#pragma unroll
    for (int r = 0; r < 8; ++r) cs[j][r] = 0.0f;

  const u16* hrow = hlds + (mt * 16 + lm) * HDIM;

  for (int s = 0; s < LEN; ++s) {
    int t = dir ? (LEN - 1 - s) : s;

    // Prefetch next step's gate fragments for this wave (16 tiles x 512B).
    if (s + 1 < LEN) {
      int tn = dir ? (LEN - 2 - s) : (s + 1);
      const u16* gnext = gates + ((size_t)(tn * 16 + mt) * 32) * 256 + (size_t)lane * 8;
#pragma unroll
      for (int g = 0; g < 4; ++g)
#pragma unroll
        for (int j = 0; j < 4; ++j)
          __builtin_prefetch(gnext + (size_t)(g * 8 + ntb + j) * 256, 0, 3);
    }

    // A fragments of h for this wave's M-tile (shared by all 4 h-tiles)
    v16bf a[4];
#pragma unroll
    for (int ks = 0; ks < 4; ++ks) {
      int ao = ks * 32 + (hi << 3);
      a[ks] = load_frag(hrow + ao, hrow + ao + 16);
    }

    float hv[4][8];                 // new h (fp32) for the 4 tiles
    const u16* gbase = gates + ((size_t)(t * 16 + mt) * 32) * 256 + (size_t)lane * 8;

#pragma unroll
    for (int j = 0; j < 4; ++j) {
      int nt = ntb + j;
      v8f acc[4];
#pragma unroll
      for (int g = 0; g < 4; ++g) {
        // init accumulator from fragment-ordered input gates: one b128/lane
        uint4 gq = *(const uint4*)(gbase + (size_t)(g * 8 + nt) * 256);
        v8f A_;
        A_[0] = bf2f((u16)(gq.x & 0xFFFFu)); A_[1] = bf2f((u16)(gq.x >> 16));
        A_[2] = bf2f((u16)(gq.y & 0xFFFFu)); A_[3] = bf2f((u16)(gq.y >> 16));
        A_[4] = bf2f((u16)(gq.z & 0xFFFFu)); A_[5] = bf2f((u16)(gq.z >> 16));
        A_[6] = bf2f((u16)(gq.w & 0xFFFFu)); A_[7] = bf2f((u16)(gq.w >> 16));
        const u16* brow = wlds + (g * HDIM + nt * 16 + lm) * HDIM;
#pragma unroll
        for (int ks = 0; ks < 4; ++ks) {
          int bo = ks * 32 + (hi << 3);
          v16bf b = load_frag(brow + bo, brow + bo + 16);
          A_ = __builtin_amdgcn_wmma_f32_16x16x32_bf16(false, a[ks], false, b, (short)0, A_, false, false);
        }
        acc[g] = A_;
      }
      // gate order i,f,g,o; c' = sig(f)*c + sig(i)*tanh(g); h = sig(o)*tanh(c')
      float* hsrow = hs + ((size_t)t * BATCH + mt * 16) * HDIM + nt * 16 + lm;
#pragma unroll
      for (int r = 0; r < 8; ++r) {
        float iv = sigm(acc[0][r]);
        float fv = sigm(acc[1][r]);
        float gv = tanhf(acc[2][r]);
        float ov = sigm(acc[3][r]);
        float c = fv * cs[j][r] + iv * gv;
        cs[j][r] = c;
        float h = ov * tanhf(c);
        hv[j][r] = h;
        hsrow[(size_t)((hi ? 8 : 0) + r) * HDIM] = h;
      }
    }

    __syncthreads();                // all h_lds reads of step s done
#pragma unroll
    for (int j = 0; j < 4; ++j) {
      int colbase = (ntb + j) * 16 + lm;
#pragma unroll
      for (int r = 0; r < 8; ++r) {
        int row = mt * 16 + (hi ? 8 : 0) + r;
        hlds[row * HDIM + colbase] = f2bf(hv[j][r]);
      }
    }
    __syncthreads();                // h_lds ready for step s+1
  }
}

// ---------------------------------------------------------------------------
// Phase 5: emissions em[t][b][c] = b_fc[c] + hs_f . w_fc[c,0:128] + hs_b . w_fc[c,128:256]
// ---------------------------------------------------------------------------
__global__ void emissions_kernel(const float* __restrict__ hs_f, const float* __restrict__ hs_b,
                                 const float* __restrict__ w_fc, const float* __restrict__ b_fc,
                                 float* __restrict__ em) {
  int idx = blockIdx.x * blockDim.x + threadIdx.x;   // L*B*9 = 1179648
  int c  = idx % NTAG;
  int lb = idx / NTAG;                               // t*B + b
  const float* hf = hs_f + (size_t)lb * HDIM;
  const float* hb = hs_b + (size_t)lb * HDIM;
  const float* wc = w_fc + (size_t)c * 256;
  float s = b_fc[c];
#pragma unroll 4
  for (int k = 0; k < HDIM; ++k) s += hf[k] * wc[k];
#pragma unroll 4
  for (int k = 0; k < HDIM; ++k) s += hb[k] * wc[HDIM + k];
  em[idx] = s;
}

// ---------------------------------------------------------------------------
// Phase 6: CRF NLL. One thread per batch element; LDS reduction to scalar.
// ---------------------------------------------------------------------------
__global__ __launch_bounds__(256)
void crf_kernel(const int* __restrict__ x, const int* __restrict__ tags,
                const float* __restrict__ em, const float* __restrict__ start,
                const float* __restrict__ endt, const float* __restrict__ trans,
                float* __restrict__ out) {
  __shared__ float red[BATCH];
  __shared__ float tr[NTAG * NTAG], st[NTAG], en[NTAG];
  int b = threadIdx.x;
  if (b < NTAG * NTAG) tr[b] = trans[b];
  if (b < NTAG) { st[b] = start[b]; en[b] = endt[b]; }
  __syncthreads();

  const int* xb = x + (size_t)b * LEN;
  const int* tb = tags + (size_t)b * LEN;

  float alpha[NTAG];
  int t0 = tb[0];
  float num = st[t0] + em[(size_t)b * NTAG + t0];
  for (int c = 0; c < NTAG; ++c) alpha[c] = st[c] + em[(size_t)b * NTAG + c];
  int seqlen = (xb[0] != 0) ? 1 : 0;

  for (int t = 1; t < LEN; ++t) {
    bool m = (xb[t] != 0);
    const float* emt = em + ((size_t)t * BATCH + b) * NTAG;
    int cur = tb[t], prev = tb[t - 1];
    if (m) { num += tr[prev * NTAG + cur] + emt[cur]; seqlen++; }
    float na[NTAG];
    for (int c2 = 0; c2 < NTAG; ++c2) {
      float mx = -1e30f;
      for (int c1 = 0; c1 < NTAG; ++c1) mx = fmaxf(mx, alpha[c1] + tr[c1 * NTAG + c2]);
      float ss = 0.0f;
      for (int c1 = 0; c1 < NTAG; ++c1) ss += expf(alpha[c1] + tr[c1 * NTAG + c2] - mx);
      na[c2] = mx + logf(ss) + emt[c2];
    }
    if (m) for (int c = 0; c < NTAG; ++c) alpha[c] = na[c];
  }
  int li = (seqlen > 0) ? (seqlen - 1) : 0;
  num += en[tb[li]];

  float mx = -1e30f;
  for (int c = 0; c < NTAG; ++c) mx = fmaxf(mx, alpha[c] + en[c]);
  float ss = 0.0f;
  for (int c = 0; c < NTAG; ++c) ss += expf(alpha[c] + en[c] - mx);
  float logZ = mx + logf(ss);

  red[b] = num - logZ;
  __syncthreads();
  if (b == 0) {
    float s = 0.0f;
    for (int i = 0; i < BATCH; ++i) s += red[i];
    out[0] = -s;
  }
}

// ---------------------------------------------------------------------------
// Host launcher
// ---------------------------------------------------------------------------
extern "C" void kernel_launch(void* const* d_in, const int* in_sizes, int n_in,
                              void* d_out, int out_size, void* d_ws, size_t ws_size,
                              hipStream_t stream) {
  (void)in_sizes; (void)n_in; (void)out_size; (void)ws_size;

  const int*   x        = (const int*)d_in[0];
  const int*   tags     = (const int*)d_in[1];
  const float* emb      = (const float*)d_in[2];
  const float* w_ih_f   = (const float*)d_in[3];
  const float* w_hh_f   = (const float*)d_in[4];
  const float* b_f      = (const float*)d_in[5];
  const float* w_ih_b   = (const float*)d_in[6];
  const float* w_hh_b   = (const float*)d_in[7];
  const float* b_b      = (const float*)d_in[8];
  const float* w_fc     = (const float*)d_in[9];
  const float* b_fc     = (const float*)d_in[10];
  const float* startT   = (const float*)d_in[11];
  const float* endT     = (const float*)d_in[12];
  const float* trans    = (const float*)d_in[13];

  char* ws = (char*)d_ws;
  const size_t NLB   = (size_t)LEN * BATCH;            // 131072
  const size_t OFF_EMB  = 0;                           // (L,B,E) bf16  = 33.5MB
  const size_t OFF_WIHF = OFF_EMB  + NLB * EDIM * 2;
  const size_t OFF_WHHF = OFF_WIHF + (size_t)GDIM * EDIM * 2;
  const size_t OFF_WIHB = OFF_WHHF + (size_t)GDIM * HDIM * 2;
  const size_t OFF_WHHB = OFF_WIHB + (size_t)GDIM * EDIM * 2;
  const size_t OFF_GF   = OFF_WHHB + (size_t)GDIM * HDIM * 2;
  const size_t OFF_GB   = OFF_GF   + NLB * GDIM * 2;   // 134MB each (fragment order)
  const size_t OFF_HSF  = OFF_GB   + NLB * GDIM * 2;
  const size_t OFF_HSB  = OFF_HSF  + NLB * HDIM * 4;   // 67MB each
  const size_t OFF_EM   = OFF_HSB  + NLB * HDIM * 4;   // 4.7MB

  u16*   embeds  = (u16*)(ws + OFF_EMB);
  u16*   wihf_bf = (u16*)(ws + OFF_WIHF);
  u16*   whhf_bf = (u16*)(ws + OFF_WHHF);
  u16*   wihb_bf = (u16*)(ws + OFF_WIHB);
  u16*   whhb_bf = (u16*)(ws + OFF_WHHB);
  u16*   gates_f = (u16*)(ws + OFF_GF);
  u16*   gates_b = (u16*)(ws + OFF_GB);
  float* hs_f    = (float*)(ws + OFF_HSF);
  float* hs_b    = (float*)(ws + OFF_HSB);
  float* em      = (float*)(ws + OFF_EM);

  const int WN = GDIM * EDIM;                          // 65536 per weight matrix
  cvt_bf16_kernel<<<WN / 256, 256, 0, stream>>>(w_ih_f, wihf_bf, WN);
  cvt_bf16_kernel<<<WN / 256, 256, 0, stream>>>(w_hh_f, whhf_bf, WN);
  cvt_bf16_kernel<<<WN / 256, 256, 0, stream>>>(w_ih_b, wihb_bf, WN);
  cvt_bf16_kernel<<<WN / 256, 256, 0, stream>>>(w_hh_b, whhb_bf, WN);

  embed_kernel<<<(int)(NLB * EDIM / 256), 256, 0, stream>>>(x, emb, embeds);

  // 262144 tiles / 8 waves per block
  gemm_gates_kernel<<<32768, 256, 0, stream>>>(embeds, wihf_bf, b_f, gates_f);
  gemm_gates_kernel<<<32768, 256, 0, stream>>>(embeds, wihb_bf, b_b, gates_b);

  const size_t lds_bytes = (size_t)(GDIM * HDIM + BATCH * HDIM) * 2;  // 196608
  lstm_rec_kernel<<<2, 1024, lds_bytes, stream>>>(whhf_bf, whhb_bf, gates_f, gates_b, hs_f, hs_b);

  emissions_kernel<<<(int)(NLB * NTAG / 256), 256, 0, stream>>>(hs_f, hs_b, w_fc, b_fc, em);

  crf_kernel<<<1, 256, 0, stream>>>(x, tags, em, startT, endT, trans, (float*)d_out);
}
